// AttentionAggregator_50869592654950
// MI455X (gfx1250) — compile-verified, matrix operand order checked by reading, not run on hardware
//
#include <hip/hip_runtime.h>

typedef _Float16 half_t;
typedef __attribute__((ext_vector_type(16))) _Float16 v16h;
typedef __attribute__((ext_vector_type(8)))  _Float16 v8h;
typedef __attribute__((ext_vector_type(4)))  _Float16 v4h;
typedef __attribute__((ext_vector_type(8)))  float    v8f;

#define WAVES   8      // waves per workgroup == nodes per workgroup
#define K_NEIB  32
#define D_IN    128
#define H_ATT   32
#define O_OUT   128
#define NEIB_LD 136    // f16 row stride: 272B -> rows stay 16B-aligned for b128 LDS ops
#define HID_LD  40     // 80B rows, 16B-aligned

// ---------------- fast tanh: native v_tanh_f32 on gfx1250 ----------------
__device__ __forceinline__ float fast_tanh(float v) {
#if __has_builtin(__builtin_amdgcn_tanhf)
    return __builtin_amdgcn_tanhf(v);
#elif __has_builtin(__builtin_amdgcn_tanh_f32)
    return __builtin_amdgcn_tanh_f32(v);
#else
    const float e = __expf(2.f * v);
    return 1.f - 2.f / (e + 1.f);
#endif
}

// ---- WMMA fragment loaders (CDNA5 16x16x32 f16 layouts, wave32) ----
// Lane's 16 halfs are two contiguous 16B runs -> 2x b128 loads when aligned.

// A: 16x32 f16 tile, row-major source, stride ld (halfs). ld*2 % 16 == 0 required.
__device__ __forceinline__ v16h load_a16(const half_t* base, int ld, int r0, int k0, int lane) {
    const int m  = lane & 15;
    const int kb = (lane < 16) ? 0 : 8;
    const half_t* p = base + (r0 + m) * ld + k0 + kb;
    const v8h lo = *(const v8h*)p;
    const v8h hi = *(const v8h*)(p + 16);
    return __builtin_shufflevector(lo, hi, 0,1,2,3,4,5,6,7,8,9,10,11,12,13,14,15);
}

// B: 32x16 tile where the SOURCE IS THE TRANSPOSE (N x K row-major f16, stride ld).
// B(k,n) = srcT[n][k] -> contiguous in k -> 2x b128 loads. Works for LDS or global.
__device__ __forceinline__ v16h load_bT16(const half_t* baseT, int ld, int k0, int n0, int lane) {
    const int n  = lane & 15;
    const int kb = (lane < 16) ? 0 : 8;
    const half_t* p = baseT + (n0 + n) * ld + k0 + kb;
    const v8h lo = *(const v8h*)p;
    const v8h hi = *(const v8h*)(p + 16);
    return __builtin_shufflevector(lo, hi, 0,1,2,3,4,5,6,7,8,9,10,11,12,13,14,15);
}

// B: 32x16 tile converted on the fly from row-major f32 global [K x N] (fallback path)
__device__ __forceinline__ v16h load_b_gf32(const float* base, int ld, int k0, int n0, int lane) {
    const int n  = lane & 15;
    const int kb = (lane < 16) ? 0 : 8;
    const float* p = base + (size_t)(k0 + kb) * ld + n0 + n;
    v16h b;
#pragma unroll
    for (int i = 0; i < 8; ++i) { b[i] = (half_t)p[(size_t)i * ld]; b[i + 8] = (half_t)p[(size_t)(16 + i) * ld]; }
    return b;
}

__device__ __forceinline__ v8f wmma_f16(v16h a, v16h b, v8f c) {
    return __builtin_amdgcn_wmma_f32_16x16x32_f16(false, a, false, b, (short)0, c, false, false);
}

// ---------------- prologue: W_x / W_neib -> transposed f16 in scratch ----------------
__global__ void transpose_w_kernel(const float* __restrict__ W_x,
                                   const float* __restrict__ W_neib,
                                   half_t* __restrict__ wT) {
    const int i = blockIdx.x * blockDim.x + threadIdx.x;   // over D_IN*O_OUT
    if (i < D_IN * O_OUT) {
        const int k = i / O_OUT, n = i % O_OUT;            // W[k][n]
        wT[(size_t)n * D_IN + k]           = (half_t)W_x[i];
        wT[(size_t)(O_OUT + n) * D_IN + k] = (half_t)W_neib[i];
    }
}

__global__ __launch_bounds__(WAVES * 32)
void attn_agg_kernel(const float* __restrict__ x, const float* __restrict__ neibs,
                     const float* __restrict__ W_att1, const float* __restrict__ W_att2,
                     const float* __restrict__ W_x, const float* __restrict__ W_neib,
                     const half_t* __restrict__ WT,   // transposed f16 [W_xT | W_neibT], or null
                     float* __restrict__ out, int n_nodes) {
    const int tid  = threadIdx.x;
    const int lane = tid & 31;
    const int wave = tid >> 5;
    int node = blockIdx.x * WAVES + wave;
    if (node >= n_nodes) node = n_nodes - 1;   // 50000 % 8 == 0 -> never taken

    __shared__ __align__(16) half_t s_w1T [H_ATT][D_IN];            // W_att1^T  8 KB
    __shared__ __align__(16) half_t s_w2T [H_ATT][H_ATT];           // W_att2^T  2 KB
    __shared__ __align__(16) half_t s_neib[WAVES][K_NEIB][NEIB_LD]; // 68 KB
    __shared__ __align__(16) half_t s_hid [WAVES][K_NEIB][HID_LD];  // 20 KB (reused for att)
    __shared__ __align__(16) float  s_xrow[WAVES][D_IN];            // 4 KB
    __shared__ __align__(16) half_t s_xh  [16][NEIB_LD];            // 4.25 KB (rows 8..15 zero)
    __shared__ __align__(16) half_t s_agg [16][NEIB_LD];            // 4.25 KB (rows 8..15 zero)
    __shared__ float  s_xhid[WAVES][H_ATT];                         // 1 KB
    __shared__ float  s_xatt[WAVES][H_ATT];                         // 1 KB
    __shared__ float  s_ws  [WAVES][K_NEIB];                        // 1 KB

    // ---- phase 0: stage attention weights TRANSPOSED (f16) + zero pad rows ----
    for (int i = tid; i < D_IN * H_ATT; i += blockDim.x)
        s_w1T[i & 31][i >> 5] = (half_t)W_att1[i];                  // W1[d][h] -> s_w1T[h][d]
    for (int i = tid; i < H_ATT * H_ATT; i += blockDim.x)
        s_w2T[i & 31][i >> 5] = (half_t)W_att2[i];
    for (int i = tid; i < 8 * NEIB_LD; i += blockDim.x) {
        s_xh [8 + i / NEIB_LD][i % NEIB_LD] = (half_t)0.f;
        s_agg[8 + i / NEIB_LD][i % NEIB_LD] = (half_t)0.f;
    }
    __syncthreads();

    // ---- phase 1: stage this node's 32x128 neighbor tile + x row ----
    {
        const float* nb = neibs + (size_t)node * K_NEIB * D_IN;
        __builtin_prefetch(nb + (size_t)WAVES * K_NEIB * D_IN, 0, 1);
        const int c0 = lane * 4;
#pragma unroll 4
        for (int r = 0; r < K_NEIB; ++r) {
            const float4 v = *(const float4*)(nb + r * D_IN + c0);
            v4h h; h[0] = (half_t)v.x; h[1] = (half_t)v.y; h[2] = (half_t)v.z; h[3] = (half_t)v.w;
            *(v4h*)&s_neib[wave][r][c0] = h;                        // ds_store_b64
        }
        const float4 xv = *(const float4*)(x + (size_t)node * D_IN + c0);
        s_xrow[wave][c0 + 0] = xv.x; s_xrow[wave][c0 + 1] = xv.y;
        s_xrow[wave][c0 + 2] = xv.z; s_xrow[wave][c0 + 3] = xv.w;
        v4h h; h[0] = (half_t)xv.x; h[1] = (half_t)xv.y; h[2] = (half_t)xv.z; h[3] = (half_t)xv.w;
        *(v4h*)&s_xh[wave][c0] = h;
    }

    // ---- phase 2: GEMM1  hidden(32x32) = tanh(neib(32x128) @ W_att1(128x32)) ----
    // B fragments preloaded once (reused across both M strips); A loaded once per
    // (mt,kt) and fed to both N tiles. mt kept rolled to bound register pressure.
    {
        const half_t* nb_lds = &s_neib[wave][0][0];
        v16h bfrag[4][2];
#pragma unroll
        for (int kt = 0; kt < 4; ++kt) {
            bfrag[kt][0] = load_bT16(&s_w1T[0][0], D_IN, kt * 32, 0,  lane);
            bfrag[kt][1] = load_bT16(&s_w1T[0][0], D_IN, kt * 32, 16, lane);
        }
#pragma unroll 1
        for (int mt = 0; mt < 2; ++mt) {
            v8f c0 = {}, c1 = {};
#pragma unroll
            for (int kt = 0; kt < 4; ++kt) {
                const v16h a = load_a16(nb_lds, NEIB_LD, mt * 16, kt * 32, lane);
                c0 = wmma_f16(a, bfrag[kt][0], c0);
                c1 = wmma_f16(a, bfrag[kt][1], c1);
            }
            const int m0 = mt * 16 + ((lane < 16) ? 0 : 8);
            const int n  = lane & 15;
#pragma unroll
            for (int r = 0; r < 8; ++r) {
                s_hid[wave][m0 + r][n]      = (half_t)fast_tanh(c0[r]);
                s_hid[wave][m0 + r][16 + n] = (half_t)fast_tanh(c1[r]);
            }
        }
    }

    // ---- phase 3: x_att (1 row, VALU; lane == h) ----
    {
        float hx = 0.f;
        const half_t* w1r = &s_w1T[lane][0];
#pragma unroll
        for (int d0 = 0; d0 < D_IN; d0 += 8) {
            const v8h w = *(const v8h*)(w1r + d0);
            const float4 xa_ = *(const float4*)&s_xrow[wave][d0];
            const float4 xb_ = *(const float4*)&s_xrow[wave][d0 + 4];
            hx += xa_.x * (float)w[0] + xa_.y * (float)w[1] + xa_.z * (float)w[2] + xa_.w * (float)w[3]
                + xb_.x * (float)w[4] + xb_.y * (float)w[5] + xb_.z * (float)w[6] + xb_.w * (float)w[7];
        }
        s_xhid[wave][lane] = fast_tanh(hx);
        float xa = 0.f;
        const half_t* w2r = &s_w2T[lane][0];
#pragma unroll
        for (int j0 = 0; j0 < H_ATT; j0 += 8) {
            const v8h w = *(const v8h*)(w2r + j0);
#pragma unroll
            for (int j = 0; j < 8; ++j) xa += s_xhid[wave][j0 + j] * (float)w[j];
        }
        s_xatt[wave][lane] = xa;
    }

    // ---- phase 4: GEMM2  att(32x32) = hidden @ W_att2 ; store att back into s_hid ----
    {
        v16h a0 = load_a16(&s_hid[wave][0][0], HID_LD, 0,  0, lane);
        v16h a1 = load_a16(&s_hid[wave][0][0], HID_LD, 16, 0, lane);
        v16h b0 = load_bT16(&s_w2T[0][0], H_ATT, 0, 0,  lane);
        v16h b1 = load_bT16(&s_w2T[0][0], H_ATT, 0, 16, lane);
        v8f c00 = {}, c01 = {}, c10 = {}, c11 = {};
        c00 = wmma_f16(a0, b0, c00);
        c01 = wmma_f16(a0, b1, c01);
        c10 = wmma_f16(a1, b0, c10);
        c11 = wmma_f16(a1, b1, c11);
        const int m0 = (lane < 16) ? 0 : 8;
        const int n  = lane & 15;
#pragma unroll
        for (int r = 0; r < 8; ++r) {
            s_hid[wave][m0 + r][n]           = (half_t)c00[r];
            s_hid[wave][m0 + r][16 + n]      = (half_t)c01[r];
            s_hid[wave][16 + m0 + r][n]      = (half_t)c10[r];
            s_hid[wave][16 + m0 + r][16 + n] = (half_t)c11[r];
        }
    }

    // ---- phase 5: scores + softmax over K (lane == neighbor) ----
    {
        float sc = 0.f;
#pragma unroll
        for (int h0 = 0; h0 < H_ATT; h0 += 8) {
            const v8h av = *(const v8h*)&s_hid[wave][lane][h0];
#pragma unroll
            for (int h = 0; h < 8; ++h) sc += (float)av[h] * s_xatt[wave][h0 + h];
        }
        float mx = sc;
#pragma unroll
        for (int off = 16; off > 0; off >>= 1) mx = fmaxf(mx, __shfl_xor(mx, off, 32));
        const float e = __expf(sc - mx);
        float sum = e;
#pragma unroll
        for (int off = 16; off > 0; off >>= 1) sum += __shfl_xor(sum, off, 32);
        s_ws[wave][lane] = e / sum;
    }

    // ---- phase 6: agg[d] = sum_k ws[k] * neib[k][d]  (from LDS tile) ----
    {
        const int c0 = lane * 4;
        float a0 = 0.f, a1 = 0.f, a2 = 0.f, a3 = 0.f;
#pragma unroll 8
        for (int k = 0; k < K_NEIB; ++k) {
            const float wk = s_ws[wave][k];
            const v4h nv = *(const v4h*)&s_neib[wave][k][c0];       // ds_load_b64
            a0 += wk * (float)nv[0];
            a1 += wk * (float)nv[1];
            a2 += wk * (float)nv[2];
            a3 += wk * (float)nv[3];
        }
        v4h h; h[0] = (half_t)a0; h[1] = (half_t)a1; h[2] = (half_t)a2; h[3] = (half_t)a3;
        *(v4h*)&s_agg[wave][c0] = h;
    }
    __syncthreads();

    // ---- phase 7: out = relu(cat(x@W_x, agg@W_neib)); wave w owns 16-col tile w of each ----
    {
        v8f c = {};
        if (WT) {
#pragma unroll
            for (int kt = 0; kt < 4; ++kt)
                c = wmma_f16(load_a16(&s_xh[0][0], NEIB_LD, 0, kt * 32, lane),
                             load_bT16(WT, D_IN, kt * 32, wave * 16, lane), c);
        } else {
#pragma unroll
            for (int kt = 0; kt < 4; ++kt)
                c = wmma_f16(load_a16(&s_xh[0][0], NEIB_LD, 0, kt * 32, lane),
                             load_b_gf32(W_x, O_OUT, kt * 32, wave * 16, lane), c);
        }
        if (lane < 16) {
            const int n = wave * 16 + lane;
#pragma unroll
            for (int r = 0; r < 8; ++r) {
                const int row = blockIdx.x * WAVES + r;
                if (row < n_nodes) out[(size_t)row * (2 * O_OUT) + n] = fmaxf(c[r], 0.f);
            }
        }
    }
    {
        v8f c = {};
        if (WT) {
            const half_t* WnT = WT + (size_t)O_OUT * D_IN;
#pragma unroll
            for (int kt = 0; kt < 4; ++kt)
                c = wmma_f16(load_a16(&s_agg[0][0], NEIB_LD, 0, kt * 32, lane),
                             load_bT16(WnT, D_IN, kt * 32, wave * 16, lane), c);
        } else {
#pragma unroll
            for (int kt = 0; kt < 4; ++kt)
                c = wmma_f16(load_a16(&s_agg[0][0], NEIB_LD, 0, kt * 32, lane),
                             load_b_gf32(W_neib, O_OUT, kt * 32, wave * 16, lane), c);
        }
        if (lane < 16) {
            const int n = O_OUT + wave * 16 + lane;
#pragma unroll
            for (int r = 0; r < 8; ++r) {
                const int row = blockIdx.x * WAVES + r;
                if (row < n_nodes) out[(size_t)row * (2 * O_OUT) + n] = fmaxf(c[r], 0.f);
            }
        }
    }
}

extern "C" void kernel_launch(void* const* d_in, const int* in_sizes, int n_in,
                              void* d_out, int out_size, void* d_ws, size_t ws_size,
                              hipStream_t stream) {
    const float* x      = (const float*)d_in[0];
    const float* neibs  = (const float*)d_in[1];
    const float* W_att1 = (const float*)d_in[2];
    const float* W_att2 = (const float*)d_in[3];
    const float* W_x    = (const float*)d_in[4];
    const float* W_neib = (const float*)d_in[5];
    float* out = (float*)d_out;

    const int N = in_sizes[0] / D_IN;                 // 50000
    const int grid = (N + WAVES - 1) / WAVES;         // 6250 (exact)

    const size_t wt_bytes = (size_t)2 * D_IN * O_OUT * sizeof(half_t);  // 64 KB
    half_t* WT = nullptr;
    if (ws_size >= wt_bytes) {
        WT = (half_t*)d_ws;
        transpose_w_kernel<<<(D_IN * O_OUT + 255) / 256, 256, 0, stream>>>(W_x, W_neib, WT);
    }
    attn_agg_kernel<<<grid, WAVES * 32, 0, stream>>>(x, neibs, W_att1, W_att2,
                                                     W_x, W_neib, WT, out, N);
}
